// Attention_17325898072594
// MI455X (gfx1250) — compile-verified, matrix operand order checked by reading, not run on hardware
//
#include <hip/hip_runtime.h>
#include <hip/hip_bf16.h>
#include <stdint.h>

typedef __attribute__((ext_vector_type(16))) __bf16 v16bf;
typedef __attribute__((ext_vector_type(8)))  __bf16 v8bf;
typedef __attribute__((ext_vector_type(4)))  __bf16 v4bf;
typedef __attribute__((ext_vector_type(8)))  float  v8f;
typedef __attribute__((ext_vector_type(4)))  uint32_t v4u;
typedef __attribute__((ext_vector_type(8)))  int     v8i;
typedef __attribute__((ext_vector_type(4)))  int     v4i;

#define D_MODEL  1024
#define N_HEADS  16
#define HEAD_DIM 64
#define BATCH    4
#define SEQ      2048
#define M_TOTAL  (BATCH * SEQ)   // 8192
#define WELEMS   (1024 * 1024)

__device__ __forceinline__ v16bf cat8(v8bf lo, v8bf hi) {
    return __builtin_shufflevector(lo, hi, 0,1,2,3,4,5,6,7,8,9,10,11,12,13,14,15);
}

__device__ __forceinline__ v8f wmma_bf(v16bf a, v16bf b, v8f c) {
    return __builtin_amdgcn_wmma_f32_16x16x32_bf16(false, a, false, b, (short)0, c, false, false);
}

// ---------------------------------------------------------------------------
// TDM: 2D tile load global->LDS via Tensor DMA Descriptor (D# groups 0/1,
// remaining groups zero => 2D). data_size=2B (bf16). One instruction moves
// the whole tile; completion tracked with TENSORcnt.
// (6-arg builtin form for this toolchain: g0, g1, g2, g3, g4, cpol.)
// ---------------------------------------------------------------------------
__device__ __forceinline__ void tdm_load_2d_bf16(
    uint32_t lds_off, const void* gaddr,
    uint32_t tensor_d0, uint32_t tensor_d1, uint32_t stride_d0,
    uint32_t tile_d0, uint32_t tile_d1)
{
    uint64_t ga = (uint64_t)(uintptr_t)gaddr;
    v4u g0;
    g0[0] = 1u;                                  // count=1 valid user descriptor
    g0[1] = lds_off;                             // LDS byte address
    g0[2] = (uint32_t)ga;                        // global_addr[31:0]
    g0[3] = (uint32_t)(ga >> 32) | (2u << 30);   // global_addr[56:32] | type=2
    v8i g1;
    g1[0] = (int)(1u << 16);                     // data_size=1 (2 bytes)
    g1[1] = (int)((tensor_d0 & 0xFFFFu) << 16);  // tensor_dim0[15:0]
    g1[2] = (int)(((tensor_d0 >> 16) & 0xFFFFu) | ((tensor_d1 & 0xFFFFu) << 16));
    g1[3] = (int)(((tensor_d1 >> 16) & 0xFFFFu) | (tile_d0 << 16));
    g1[4] = (int)tile_d1;                        // tile_dim1 | tile_dim2=0
    g1[5] = (int)stride_d0;                      // tensor_dim0_stride[31:0]
    g1[6] = 0;                                   // stride0 hi | stride1 lo
    g1[7] = 0;
    v4i z4 = {0, 0, 0, 0};
    v8i z8 = {0, 0, 0, 0, 0, 0, 0, 0};
    __builtin_amdgcn_tensor_load_to_lds(g0, g1, z4, z4, z8, 0);
}

__device__ __forceinline__ uint32_t lds_offset_of(const void* p) {
    return (uint32_t)(uintptr_t)p;   // generic LDS addr: low 32 bits = LDS offset
}

// ---------------------------------------------------------------------------
// Kernel 0: one-shot precision/layout conversion.
//  x [8192,1024] f32 -> xb bf16 (same layout), vectorized float4.
//  Wq/Wk/Wv/Wo [1024k,1024n] f32 -> bf16 transposed n-major [n][k] so B
//  fragments (and TDM tiles) are k-contiguous.
// ---------------------------------------------------------------------------
__global__ __launch_bounds__(256) void convert_kernel(
    const float* __restrict__ x,
    const float* __restrict__ Wq, const float* __restrict__ Wk,
    const float* __restrict__ Wv, const float* __restrict__ Wo,
    __bf16* __restrict__ xb, __bf16* __restrict__ wtb, __bf16* __restrict__ wob)
{
    const size_t XV = (size_t)M_TOTAL * D_MODEL / 4;       // 2097152 float4 tasks
    const size_t NT = XV + 4 * (size_t)WELEMS;             // + 4 transposes
    const size_t stride = (size_t)gridDim.x * blockDim.x;
    for (size_t i = blockIdx.x * (size_t)blockDim.x + threadIdx.x; i < NT; i += stride) {
        if (i < XV) {
            float4 f = ((const float4*)x)[i];
            v4bf o = { (__bf16)f.x, (__bf16)f.y, (__bf16)f.z, (__bf16)f.w };
            *(v4bf*)&xb[i * 4] = o;
        } else {
            size_t r = i - XV;
            int w = (int)(r >> 20);
            int e = (int)(r & (WELEMS - 1));
            int k = e >> 10, n = e & 1023;                 // read coalesced along n
            const float* W = (w == 0) ? Wq : (w == 1) ? Wk : (w == 2) ? Wv : Wo;
            __bf16* dst = (w < 3) ? (wtb + (size_t)w * WELEMS) : wob;
            dst[(size_t)n * 1024 + k] = (__bf16)W[(size_t)k * 1024 + n];
        }
    }
}

// ---------------------------------------------------------------------------
// Kernel 1: QKV projection, TDM-staged. xb[8192,1024]bf16 x Wt[1024n,1024k]
// -> Q (pre-scaled 1/8), K as [B,H,S,Dh]; V transposed [B,H,Dh,S].
// Grid: 3 * 512 mtiles * 2 nhalves = 3072 blocks, 256 threads (8 waves).
// ---------------------------------------------------------------------------
__global__ __launch_bounds__(256) void qkv_kernel(
    const __bf16* __restrict__ xb, const __bf16* __restrict__ wtb,
    __bf16* __restrict__ q_ws, __bf16* __restrict__ k_ws,
    __bf16* __restrict__ vt_ws)
{
    __shared__ __bf16 lds_a[16 * 32];
    __shared__ __bf16 lds_b[512 * 32];

    const int tid  = threadIdx.x;
    const int wid  = tid >> 5;
    const int lane = tid & 31;

    const int mat   = blockIdx.x / 1024;          // 0=Q 1=K 2=V
    const int rem   = blockIdx.x % 1024;
    const int mbase = (rem >> 1) * 16;
    const int nb0   = (rem & 1) * 512;
    const __bf16* Wt = wtb + (size_t)mat * WELEMS;

    const int sel = (lane & 16) ? 8 : 0;
    const int l15 = lane & 15;
    const uint32_t lao = lds_offset_of(lds_a);
    const uint32_t lbo = lds_offset_of(lds_b);

    v8f acc[4] = {};

    for (int kk = 0; kk < 1024; kk += 32) {
        if (wid == 0) {
            // A: 16 rows x 32 k from xb (row stride 1024)
            tdm_load_2d_bf16(lao, xb + (size_t)mbase * 1024 + kk,
                             1024, M_TOTAL, 1024, 32, 16);
            // B: 512 cols x 32 k from Wt (n-major, row stride 1024)
            tdm_load_2d_bf16(lbo, Wt + (size_t)nb0 * 1024 + kk,
                             1024, 1024, 1024, 32, 512);
            __builtin_amdgcn_s_wait_tensorcnt(0);
        }
        __syncthreads();

        v16bf af = cat8(*(const v8bf*)&lds_a[l15 * 32 + sel],
                        *(const v8bf*)&lds_a[l15 * 32 + 16 + sel]);
        const int nw = wid * 64;
        #pragma unroll
        for (int t = 0; t < 4; ++t) {
            int c0 = nw + t * 16 + l15;
            v16bf bf = cat8(*(const v8bf*)&lds_b[c0 * 32 + sel],
                            *(const v8bf*)&lds_b[c0 * 32 + 16 + sel]);
            acc[t] = wmma_bf(af, bf, acc[t]);
        }
        __syncthreads();
    }

    const int rowh  = (lane >> 4) << 3;
    const float scl = (mat == 0) ? 0.125f : 1.0f;   // 1/sqrt(Dh) folded into Q
    #pragma unroll
    for (int t = 0; t < 4; ++t) {
        #pragma unroll
        for (int j = 0; j < 8; ++j) {
            int mrow = mbase + j + rowh;
            int n    = nb0 + wid * 64 + t * 16 + l15;
            int b    = mrow >> 11, s = mrow & 2047;
            int h    = n >> 6,     d = n & 63;
            float v  = acc[t][j] * scl;
            size_t bh = (size_t)(b * 16 + h);
            if (mat == 0)      q_ws[(bh * 2048 + s) * 64 + d]  = (__bf16)v;
            else if (mat == 1) k_ws[(bh * 2048 + s) * 64 + d]  = (__bf16)v;
            else               vt_ws[(bh * 64 + d) * 2048 + s] = (__bf16)v;  // V^T
        }
    }
}

// QK^T score tile: Q(16x64) x K^T(64x16) via two k=32 WMMAs.
__device__ __forceinline__ v8f qk_tile(const __bf16* kp, int sel,
                                       v16bf aq0, v16bf aq1)
{
    v16bf bk0 = cat8(*(const v8bf*)&kp[sel],      *(const v8bf*)&kp[16 + sel]);
    v16bf bk1 = cat8(*(const v8bf*)&kp[32 + sel], *(const v8bf*)&kp[48 + sel]);
    v8f c = {};
    c = wmma_bf(aq0, bk0, c);
    c = wmma_bf(aq1, bk1, c);
    return c;
}

// ---------------------------------------------------------------------------
// Kernel 2: causal attention. One wave per (b,h,16-query-row tile).
// Phase 1: WMMA QK^T + online max/sum. Phase 2: recompute, emit probs via a
// fp32 LDS transpose (coalesced b128 stores on the 1 GB probs stream), bf16
// LDS transpose feeding P x V^T WMMAs. Upper triangle zero-filled float4.
// ---------------------------------------------------------------------------
__global__ __launch_bounds__(256) void attn_kernel(
    const __bf16* __restrict__ q_ws, const __bf16* __restrict__ k_ws,
    const __bf16* __restrict__ vt_ws,
    __bf16* __restrict__ attn_ws, float* __restrict__ probs)
{
    __shared__ __bf16 pbuf[8][16][32];    // per-wave prob tile (bf16, A-layout src)
    __shared__ float  pfbuf[8][16][16];   // per-wave prob tile (f32, coalesce)

    const int tid  = threadIdx.x;
    const int wid  = tid >> 5;
    const int lane = tid & 31;

    const int task = blockIdx.x * 8 + wid;
    const int b  = task >> 11;
    const int rm = task & 2047;
    const int h  = rm >> 7;
    const int qt = rm & 127;
    const int qbase = qt * 16;

    const int sel  = (lane & 16) ? 8 : 0;
    const int l15  = lane & 15;
    const int rowh = (lane >> 4) << 3;
    const size_t bh = (size_t)(b * 16 + h);

    const __bf16* qp = q_ws + (bh * 2048 + qbase + l15) * 64;
    v16bf aq0 = cat8(*(const v8bf*)&qp[sel],      *(const v8bf*)&qp[16 + sel]);
    v16bf aq1 = cat8(*(const v8bf*)&qp[32 + sel], *(const v8bf*)&qp[48 + sel]);

    float m[8], l[8];
    #pragma unroll
    for (int j = 0; j < 8; ++j) { m[j] = -3.0e38f; l[j] = 0.0f; }

    // ---- phase 1: online row max / sum ----
    for (int jt = 0; jt <= qt; ++jt) {
        const int col = jt * 16 + l15;
        v8f c = qk_tile(k_ws + (bh * 2048 + col) * 64, sel, aq0, aq1);
        #pragma unroll
        for (int j = 0; j < 8; ++j) {
            float v = c[j];
            if (jt == qt && col > qbase + j + rowh) v = -3.0e38f;
            float rmax = v;
            rmax = fmaxf(rmax, __shfl_xor(rmax, 1, 32));
            rmax = fmaxf(rmax, __shfl_xor(rmax, 2, 32));
            rmax = fmaxf(rmax, __shfl_xor(rmax, 4, 32));
            rmax = fmaxf(rmax, __shfl_xor(rmax, 8, 32));
            float mn = fmaxf(m[j], rmax);
            float p  = __expf(v - mn);
            float rs = p;
            rs += __shfl_xor(rs, 1, 32);
            rs += __shfl_xor(rs, 2, 32);
            rs += __shfl_xor(rs, 4, 32);
            rs += __shfl_xor(rs, 8, 32);
            l[j] = l[j] * __expf(m[j] - mn) + rs;
            m[j] = mn;
        }
    }
    float invl[8];
    #pragma unroll
    for (int j = 0; j < 8; ++j) invl[j] = 1.0f / l[j];

    v8f acc[4] = {};

    // ---- phase 2: recompute, emit probs, P x V ----
    for (int jt = 0; jt <= qt; ++jt) {
        const int col = jt * 16 + l15;
        v8f c = qk_tile(k_ws + (bh * 2048 + col) * 64, sel, aq0, aq1);
        #pragma unroll
        for (int j = 0; j < 8; ++j) {
            float v = c[j];
            int row = qbase + j + rowh;
            if (jt == qt && col > row) v = -3.0e38f;
            float p = __expf(v - m[j]) * invl[j];
            pfbuf[wid][j + rowh][l15] = p;                       // f32 transpose
            pbuf[wid][j + rowh][(jt & 1) * 16 + l15] = (__bf16)p;// bf16 A-layout
        }
        asm volatile("s_wait_dscnt 0" ::: "memory");   // cross-lane LDS RAW

        // coalesced probs write: lane -> (row l15, col chunk (lane>>4)*8)
        {
            const int cchunk = (lane >> 4) * 8;
            float4 p0 = *(const float4*)&pfbuf[wid][l15][cchunk];
            float4 p1 = *(const float4*)&pfbuf[wid][l15][cchunk + 4];
            float* pr = probs + (bh * 2048 + qbase + l15) * 2048 + jt * 16 + cchunk;
            *(float4*)pr       = p0;
            *(float4*)(pr + 4) = p1;
        }

        const bool flush = (jt & 1) || (jt == qt);
        if (flush) {
            if ((jt == qt) && !(jt & 1)) {   // odd tile count: zero-pad k=16..31
                v8bf z = {};
                *(v8bf*)&pbuf[wid][l15][16 + sel] = z;
                asm volatile("s_wait_dscnt 0" ::: "memory");
            }
            v16bf ap = cat8(*(const v8bf*)&pbuf[wid][l15][sel],
                            *(const v8bf*)&pbuf[wid][l15][16 + sel]);
            const int kb = (jt & ~1) * 16;
            #pragma unroll
            for (int t = 0; t < 4; ++t) {
                const __bf16* vp = vt_ws + (bh * 64 + t * 16 + l15) * 2048 + kb;
                v16bf bv = cat8(*(const v8bf*)&vp[sel], *(const v8bf*)&vp[16 + sel]);
                acc[t] = wmma_bf(ap, bv, acc[t]);
            }
        }
    }

    // ---- zero-fill masked columns, coalesced float4 ----
    const int zstart = (qt + 1) * 16;
    if (zstart < 2048) {
        float4 z4 = make_float4(0.f, 0.f, 0.f, 0.f);
        const int n4 = (2048 - zstart) >> 2;
        for (int r = 0; r < 16; ++r) {
            float4* zp = (float4*)(probs + (bh * 2048 + qbase + r) * 2048 + zstart);
            for (int i = lane; i < n4; i += 32) zp[i] = z4;
        }
    }

    // ---- attn tile bf16, layout [B,S,H*Dh] for the output GEMM ----
    #pragma unroll
    for (int t = 0; t < 4; ++t) {
        #pragma unroll
        for (int j = 0; j < 8; ++j) {
            int s = qbase + j + rowh;
            int d = t * 16 + l15;
            attn_ws[((size_t)b * 2048 + s) * 1024 + h * 64 + d] = (__bf16)acc[t][j];
        }
    }
}

// ---------------------------------------------------------------------------
// Kernel 3: output projection, TDM-staged. attn[8192,1024]bf16 x WoT + bo.
// ---------------------------------------------------------------------------
__global__ __launch_bounds__(256) void oproj_kernel(
    const __bf16* __restrict__ attn_ws, const __bf16* __restrict__ wob,
    const float* __restrict__ bo, float* __restrict__ out)
{
    __shared__ __bf16 lds_a[16 * 32];
    __shared__ __bf16 lds_b[512 * 32];

    const int tid  = threadIdx.x;
    const int wid  = tid >> 5;
    const int lane = tid & 31;

    const int mbase = (blockIdx.x >> 1) * 16;
    const int nb0   = (blockIdx.x & 1) * 512;
    const int sel = (lane & 16) ? 8 : 0;
    const int l15 = lane & 15;
    const uint32_t lao = lds_offset_of(lds_a);
    const uint32_t lbo = lds_offset_of(lds_b);

    v8f acc[4] = {};

    for (int kk = 0; kk < 1024; kk += 32) {
        if (wid == 0) {
            tdm_load_2d_bf16(lao, attn_ws + (size_t)mbase * 1024 + kk,
                             1024, M_TOTAL, 1024, 32, 16);
            tdm_load_2d_bf16(lbo, wob + (size_t)nb0 * 1024 + kk,
                             1024, 1024, 1024, 32, 512);
            __builtin_amdgcn_s_wait_tensorcnt(0);
        }
        __syncthreads();

        v16bf af = cat8(*(const v8bf*)&lds_a[l15 * 32 + sel],
                        *(const v8bf*)&lds_a[l15 * 32 + 16 + sel]);
        const int nw = wid * 64;
        #pragma unroll
        for (int t = 0; t < 4; ++t) {
            int c0 = nw + t * 16 + l15;
            v16bf bf = cat8(*(const v8bf*)&lds_b[c0 * 32 + sel],
                            *(const v8bf*)&lds_b[c0 * 32 + 16 + sel]);
            acc[t] = wmma_bf(af, bf, acc[t]);
        }
        __syncthreads();
    }

    const int rowh = (lane >> 4) << 3;
    #pragma unroll
    for (int t = 0; t < 4; ++t) {
        #pragma unroll
        for (int j = 0; j < 8; ++j) {
            int mrow = mbase + j + rowh;
            int n    = nb0 + wid * 64 + t * 16 + l15;
            out[(size_t)mrow * 1024 + n] = acc[t][j] + bo[n];
        }
    }
}

extern "C" void kernel_launch(void* const* d_in, const int* in_sizes, int n_in,
                              void* d_out, int out_size, void* d_ws, size_t ws_size,
                              hipStream_t stream)
{
    const float* x  = (const float*)d_in[0];
    const float* Wq = (const float*)d_in[1];
    const float* Wk = (const float*)d_in[2];
    const float* Wv = (const float*)d_in[3];
    const float* Wo = (const float*)d_in[4];
    const float* bo = (const float*)d_in[5];

    float* out   = (float*)d_out;                       // [4,2048,1024]
    float* probs = out + (size_t)M_TOTAL * D_MODEL;     // [4,16,2048,2048]

    const size_t QS = (size_t)BATCH * N_HEADS * SEQ * HEAD_DIM;  // 8388608 elems
    __bf16* q_ws    = (__bf16*)d_ws;
    __bf16* k_ws    = q_ws + QS;
    __bf16* vt_ws   = k_ws + QS;
    __bf16* attn_ws = vt_ws + QS;
    __bf16* xb      = attn_ws + QS;                 // [8192,1024] bf16
    __bf16* wtb     = xb + QS;                      // [3][1024n][1024k] bf16
    __bf16* wob     = wtb + 3 * (size_t)WELEMS;     // [1024n][1024k] bf16

    convert_kernel<<<4096, 256, 0, stream>>>(x, Wq, Wk, Wv, Wo, xb, wtb, wob);
    qkv_kernel<<<3072, 256, 0, stream>>>(xb, wtb, q_ws, k_ws, vt_ws);
    attn_kernel<<<1024, 256, 0, stream>>>(q_ws, k_ws, vt_ws, attn_ws, probs);
    oproj_kernel<<<1024, 256, 0, stream>>>(attn_ws, wob, bo, out);
}
